// AttnReadout_29970281791888
// MI455X (gfx1250) — compile-verified, matrix operand order checked by reading, not run on hardware
//
#include <hip/hip_runtime.h>
#include <cstdint>

// ---------- problem constants (from reference) ----------
#define N_NODES 1000000
#define B_SEGS  20000
#define DIMD    128
#define DIMH    128
#define DIMO    256
#define BN_EPS  1e-5f

// ---------- types ----------
typedef __attribute__((ext_vector_type(16))) __bf16 v16bf;
typedef __attribute__((ext_vector_type(8)))  float  v8f;

// ---------- helpers ----------
// Native f32 -> bf16 conversion (hardware cvt on gfx1250).
__device__ __forceinline__ __bf16 tobf(float f) { return (__bf16)f; }

// Packed pair conversion: prefer the packed-convert builtin when the
// toolchain exposes it (one v_cvt_pk_bf16_f32 per 2 elements).
__device__ __forceinline__ void tobf2(float lo, float hi, __bf16* dst) {
#if defined(__has_builtin) && __has_builtin(__builtin_amdgcn_cvt_pk_bf16_f32)
  typedef __attribute__((ext_vector_type(2))) __bf16 v2bf;
  v2bf p = __builtin_amdgcn_cvt_pk_bf16_f32(lo, hi);
  dst[0] = p.x; dst[1] = p.y;
#else
  dst[0] = (__bf16)lo; dst[1] = (__bf16)hi;
#endif
}

// monotonic float->uint key for atomicMax-based segment max
__device__ __forceinline__ unsigned int fkey(float f) {
  unsigned int b = __float_as_uint(f);
  return (b & 0x80000000u) ? ~b : (b | 0x80000000u);
}
__device__ __forceinline__ float fdec(unsigned int k) {
  unsigned int b = (k & 0x80000000u) ? (k & 0x7FFFFFFFu) : ~k;
  return __uint_as_float(b);
}
__device__ __forceinline__ v8f wmma_bf16(v16bf a, v16bf b, v8f c) {
  return __builtin_amdgcn_wmma_f32_16x16x32_bf16(false, a, false, b,
                                                 (short)0, c, false, false);
}
// Build 4 A-fragments (K=0..127 in chunks of 32) for one 16-row tile,
// applying BN affine on the fly.  Per ISA A layout (16-bit, 16x32):
// lanes 0-15 hold K 0..7 (halves 0-7) and 16..23 (halves 8-15) of row M=lane;
// lanes 16-31 hold K 8..15 and 24..31 of row M=lane-16.  hi = (lane>=16)?8:0.
__device__ __forceinline__ void build_a_norm(const float* __restrict__ rp,
                                             const float* __restrict__ scale,
                                             const float* __restrict__ shift,
                                             int hi, v16bf a[4]) {
#pragma unroll
  for (int kc = 0; kc < 4; ++kc) {
    int c0 = kc * 32 + hi;
#pragma unroll
    for (int j = 0; j < 8; ++j) {
      a[kc][j]     = tobf(rp[c0 + j]      * scale[c0 + j]      + shift[c0 + j]);
      a[kc][8 + j] = tobf(rp[c0 + 16 + j] * scale[c0 + 16 + j] + shift[c0 + 16 + j]);
    }
  }
}

// ---------- K0: zero a uint32 region ----------
__global__ void k_zero(unsigned int* p, long long n) {
  long long i = (long long)blockIdx.x * blockDim.x + threadIdx.x;
  long long s = (long long)gridDim.x * blockDim.x;
  for (; i < n; i += s) p[i] = 0u;
}

// ---------- K1: BN statistics (sum / sumsq per channel) ----------
__global__ void k_bn_stats(const float* __restrict__ feat, const int* __restrict__ iid,
                           float* __restrict__ gsum /*[256]: sum[128],sumsq[128]*/) {
  __shared__ float ls[256];
  int t = threadIdx.x;
  ls[t] = 0.f;
  __syncthreads();
  int lane = t & 31, w = t >> 5;
  long long wid = (long long)blockIdx.x * 8 + w;
  long long nw  = (long long)gridDim.x * 8;
  float s0=0,s1=0,s2=0,s3=0,q0=0,q1=0,q2=0,q3=0;
  for (long long n = wid; n < N_NODES; n += nw) {
    long long row = iid[n];
    const float4* rp = (const float4*)(feat + row * DIMD);
    float4 v = rp[lane];
    s0+=v.x; s1+=v.y; s2+=v.z; s3+=v.w;
    q0+=v.x*v.x; q1+=v.y*v.y; q2+=v.z*v.z; q3+=v.w*v.w;
  }
  int c = lane * 4;
  atomicAdd(&ls[c+0], s0); atomicAdd(&ls[c+1], s1);
  atomicAdd(&ls[c+2], s2); atomicAdd(&ls[c+3], s3);
  atomicAdd(&ls[128+c+0], q0); atomicAdd(&ls[128+c+1], q1);
  atomicAdd(&ls[128+c+2], q2); atomicAdd(&ls[128+c+3], q3);
  __syncthreads();
  atomicAdd(&gsum[t], ls[t]);
}

// ---------- K2: finalize BN affine + pack weights into WMMA B-fragment layout ----------
// B-fragment pack convention (mirror of A): dest[((nt*4+kc)*32+lane)*16+h] holds
// W[k, n] with n = nt*16 + (lane&15), hi=(lane&16)?8:0, k = kc*32 + hi + (h<8 ? h : h+8).
__global__ void k_finalize_pack(const float* __restrict__ gsum,
                                const float* __restrict__ gamma, const float* __restrict__ beta,
                                const float* __restrict__ Wu, const float* __restrict__ Wv,
                                const float* __restrict__ Wout,
                                float* __restrict__ scale, float* __restrict__ shift,
                                __bf16* __restrict__ WuP,
                                __bf16* __restrict__ WvP,
                                __bf16* __restrict__ WoutP) {
  int t = threadIdx.x;
  if (t < 128) {
    float mean = gsum[t] * (1.0f / (float)N_NODES);
    float var  = gsum[128 + t] * (1.0f / (float)N_NODES) - mean * mean;
    float inv  = rsqrtf(var + BN_EPS);
    float sc   = gamma[t] * inv;
    scale[t] = sc;
    shift[t] = beta[t] - mean * sc;
  }
  for (int d = t; d < 8 * 4 * 32 * 16; d += blockDim.x) {   // Wu, Wv: 128x128
    int h = d & 15, lane = (d >> 4) & 31, kc = (d >> 9) & 3, nt = d >> 11;
    int n  = nt * 16 + (lane & 15);
    int hi = (lane & 16) ? 8 : 0;
    int k  = kc * 32 + hi + ((h < 8) ? h : (h + 8));
    WuP[d] = tobf(Wu[k * DIMH + n]);
    WvP[d] = tobf(Wv[k * DIMH + n]);
  }
  for (int d = t; d < 16 * 4 * 32 * 16; d += blockDim.x) {  // Wout: 128x256
    int h = d & 15, lane = (d >> 4) & 31, kc = (d >> 9) & 3, nt = d >> 11;
    int n  = nt * 16 + (lane & 15);
    int hi = (lane & 16) ? 8 : 0;
    int k  = kc * 32 + hi + ((h < 8) ? h : (h + 8));
    WoutP[d] = tobf(Wout[k * DIMO + n]);
  }
}

// ---------- K3: feat_v = BN(x[last_nodes]) @ W_v + b_v  (WMMA) ----------
__global__ void k_featv(const float* __restrict__ feat, const int* __restrict__ iid,
                        const int* __restrict__ last, const float* __restrict__ bv,
                        const float* __restrict__ scale, const float* __restrict__ shift,
                        const __bf16* __restrict__ WvP, float* __restrict__ featv) {
  __shared__ __align__(32) __bf16 wlds[16384];
  int t = threadIdx.x;
  {
    const uint4* src = (const uint4*)WvP;
    uint4* dst = (uint4*)wlds;
    for (int i = t; i < 2048; i += 256) dst[i] = src[i];
  }
  __syncthreads();
  int lane = t & 31, w = t >> 5;
  int g0 = blockIdx.x * 16;
  int hi = (lane >> 4) * 8;
  long long node = last[g0 + (lane & 15)];
  long long row  = iid[node];
  const float* rp = feat + row * DIMD;
  v16bf a[4];
  build_a_norm(rp, scale, shift, hi, a);
  v8f acc = {};
  const v16bf* bp = (const v16bf*)wlds;
#pragma unroll
  for (int kc = 0; kc < 4; ++kc) {
    v16bf b = bp[(w * 4 + kc) * 32 + lane];
    acc = wmma_bf16(a[kc], b, acc);
  }
  int col   = w * 16 + (lane & 15);
  int mbase = (lane >> 4) * 8;
  float bb  = bv[col];
#pragma unroll
  for (int r = 0; r < 8; ++r)
    featv[(long long)(g0 + mbase + r) * DIMH + col] = acc[r] + bb;
}

// ---------- K4: fused  e = sigmoid(BN(x)@W_u + feat_v[seg]) . w_e  + segment max ----------
__global__ void k_e(const float* __restrict__ feat, const int* __restrict__ iid,
                    const int* __restrict__ seg, const float* __restrict__ we,
                    const float* __restrict__ scale, const float* __restrict__ shift,
                    const __bf16* __restrict__ WuP, const float* __restrict__ featv,
                    float* __restrict__ e_out, unsigned int* __restrict__ emax) {
  __shared__ __align__(32) __bf16 wlds[16384];
  int t = threadIdx.x;
  {
    const uint4* src = (const uint4*)WuP;
    uint4* dst = (uint4*)wlds;
    for (int i = t; i < 2048; i += 256) dst[i] = src[i];
  }
  __syncthreads();
  int lane = t & 31, w = t >> 5;
  long long tile = (long long)blockIdx.x * 8 + w;
  if (tile * 16 >= N_NODES) return;          // no barriers past this point
  int n0 = (int)(tile * 16);
  int hi = (lane >> 4) * 8;
  long long row = iid[n0 + (lane & 15)];
  const float* rp = feat + row * DIMD;
  v16bf a[4];
  build_a_norm(rp, scale, shift, hi, a);
  int mbase = (lane >> 4) * 8;               // C/D rows this lane owns
  int segr[8];
#pragma unroll
  for (int r = 0; r < 8; ++r) segr[r] = seg[n0 + mbase + r];
  float eacc[8] = {0,0,0,0,0,0,0,0};
  const v16bf* bp = (const v16bf*)wlds;
  for (int nt = 0; nt < 8; ++nt) {           // 8 col tiles of H=128
    v8f acc = {};
#pragma unroll
    for (int kc = 0; kc < 4; ++kc) {
      v16bf b = bp[(nt * 4 + kc) * 32 + lane];
      acc = wmma_bf16(a[kc], b, acc);
    }
    int col = nt * 16 + (lane & 15);
    float wec = we[col];
#pragma unroll
    for (int r = 0; r < 8; ++r) {
      float u  = acc[r] + featv[(long long)segr[r] * DIMH + col];
      float sg = 1.0f / (1.0f + __expf(-u));
      eacc[r] += sg * wec;
    }
  }
  // reduce over the 16 columns held by each half-wave
#pragma unroll
  for (int r = 0; r < 8; ++r) {
    float v = eacc[r];
    v += __shfl_xor(v, 1, 16);
    v += __shfl_xor(v, 2, 16);
    v += __shfl_xor(v, 4, 16);
    v += __shfl_xor(v, 8, 16);
    eacc[r] = v;
  }
  if ((lane & 15) == 0) {
#pragma unroll
    for (int r = 0; r < 8; ++r) {
      e_out[n0 + mbase + r] = eacc[r];
      atomicMax(&emax[segr[r]], fkey(eacc[r]));
    }
  }
}

// ---------- K5: ex = exp(e - max[seg]); denom[seg] += ex ----------
__global__ void k_denom(const int* __restrict__ seg, const unsigned int* __restrict__ emax,
                        float* __restrict__ e, float* __restrict__ denom) {
  long long i  = (long long)blockIdx.x * blockDim.x + threadIdx.x;
  long long st = (long long)gridDim.x * blockDim.x;
  for (; i < N_NODES; i += st) {
    int s = seg[i];
    float ex = __expf(e[i] - fdec(emax[s]));
    e[i] = ex;                // overwrite e with exp-numerator
    atomicAdd(&denom[s], ex);
  }
}

// ---------- K6: rst[seg] += BN(x) * alpha  (segment-carried accumulation) ----------
__global__ void k_rst(const float* __restrict__ feat, const int* __restrict__ iid,
                      const int* __restrict__ seg, const float* __restrict__ scale,
                      const float* __restrict__ shift, const float* __restrict__ e,
                      const float* __restrict__ denom, float* __restrict__ rst) {
  int t = threadIdx.x, lane = t & 31;
  long long wid = ((long long)blockIdx.x * blockDim.x + t) >> 5;
  long long nw  = ((long long)gridDim.x * blockDim.x) >> 5;
  long long chunk = (N_NODES + nw - 1) / nw;
  long long i0 = wid * chunk;
  long long i1 = i0 + chunk; if (i1 > N_NODES) i1 = N_NODES;
  int c = lane * 4;
  float sc0=scale[c], sc1=scale[c+1], sc2=scale[c+2], sc3=scale[c+3];
  float sh0=shift[c], sh1=shift[c+1], sh2=shift[c+2], sh3=shift[c+3];
  int cur = -1;
  float a0=0,a1=0,a2=0,a3=0;
  for (long long i = i0; i < i1; ++i) {
    int s = seg[i];
    if (s != cur) {
      if (cur >= 0) {
        float* rb = rst + (long long)cur * DIMD + c;
        atomicAdd(rb+0,a0); atomicAdd(rb+1,a1); atomicAdd(rb+2,a2); atomicAdd(rb+3,a3);
      }
      cur = s; a0=a1=a2=a3=0.f;
    }
    float alpha = e[i] / denom[s];
    long long row = iid[i];
    const float4* rp = (const float4*)(feat + row * DIMD);
    float4 v = rp[lane];
    a0 += (v.x*sc0+sh0)*alpha; a1 += (v.y*sc1+sh1)*alpha;
    a2 += (v.z*sc2+sh2)*alpha; a3 += (v.w*sc3+sh3)*alpha;
  }
  if (cur >= 0) {
    float* rb = rst + (long long)cur * DIMD + c;
    atomicAdd(rb+0,a0); atomicAdd(rb+1,a1); atomicAdd(rb+2,a2); atomicAdd(rb+3,a3);
  }
}

// ---------- K7: out = rst @ W_out  (WMMA) ----------
__global__ void k_out(const float* __restrict__ rst, const __bf16* __restrict__ WoutP,
                      float* __restrict__ out) {
  int t = threadIdx.x, lane = t & 31, w = t >> 5;
  int rowTile = blockIdx.x >> 1;
  int ntg = (blockIdx.x & 1) * 8 + w;        // 16 col tiles of O=256
  int hi = (lane >> 4) * 8;
  long long g = (long long)rowTile * 16 + (lane & 15);
  const float* rp = rst + g * DIMD;
  v16bf a[4];
#pragma unroll
  for (int kc = 0; kc < 4; ++kc) {
    int c0 = kc * 32 + hi;
#pragma unroll
    for (int j = 0; j < 8; j += 2) {
      __bf16 tmp0[2], tmp1[2];
      tobf2(rp[c0 + j],      rp[c0 + j + 1],      tmp0);
      tobf2(rp[c0 + 16 + j], rp[c0 + 16 + j + 1], tmp1);
      a[kc][j]     = tmp0[0]; a[kc][j + 1]     = tmp0[1];
      a[kc][8 + j] = tmp1[0]; a[kc][8 + j + 1] = tmp1[1];
    }
  }
  v8f acc = {};
  const v16bf* bp = (const v16bf*)WoutP;
#pragma unroll
  for (int kc = 0; kc < 4; ++kc) {
    v16bf b = bp[(ntg * 4 + kc) * 32 + lane];
    acc = wmma_bf16(a[kc], b, acc);
  }
  int col   = ntg * 16 + (lane & 15);
  int mbase = (lane >> 4) * 8;
#pragma unroll
  for (int r = 0; r < 8; ++r)
    out[(long long)(rowTile * 16 + mbase + r) * DIMO + col] = acc[r];
}

// ---------- launch ----------
extern "C" void kernel_launch(void* const* d_in, const int* in_sizes, int n_in,
                              void* d_out, int out_size, void* d_ws, size_t ws_size,
                              hipStream_t stream) {
  const float* feat  = (const float*)d_in[0];
  const int*   iid   = (const int*)  d_in[1];
  const int*   seg   = (const int*)  d_in[2];
  const int*   last  = (const int*)  d_in[3];
  const float* gamma = (const float*)d_in[4];
  const float* beta  = (const float*)d_in[5];
  const float* Wu    = (const float*)d_in[6];
  const float* Wv    = (const float*)d_in[7];
  const float* bv    = (const float*)d_in[8];
  const float* we    = (const float*)d_in[9];
  const float* Wout  = (const float*)d_in[10];
  float* out = (float*)d_out;

  char* ws = (char*)d_ws;
  float*        sums  = (float*)(ws + 0);           // 256 f32
  float*        scale = (float*)(ws + 1024);        // 128 f32
  float*        shift = scale + 128;                // 128 f32
  __bf16*       WuP   = (__bf16*)(ws + 2048);       // 32 KB
  __bf16*       WvP   = (__bf16*)(ws + 34816);      // 32 KB
  __bf16*       WoutP = (__bf16*)(ws + 67584);      // 64 KB
  float*        featv = (float*)(ws + 133120);      // B*128 f32
  float*        ebuf  = (float*)(ws + 10373120);    // N f32
  unsigned int* emax  = (unsigned int*)(ws + 14373120);   // B u32
  float*        denom = (float*)(ws + 14453120);    // B f32
  float*        rst   = (float*)(ws + 14533120);    // B*128 f32  (ends ~24.8 MB)

  // re-initialize all accumulators every call (deterministic across replays)
  k_zero<<<1, 256, 0, stream>>>((unsigned int*)sums, 256);
  k_zero<<<2048, 256, 0, stream>>>(emax, (long long)(B_SEGS + B_SEGS + B_SEGS * DIMD));

  k_bn_stats<<<1024, 256, 0, stream>>>(feat, iid, sums);
  k_finalize_pack<<<1, 256, 0, stream>>>(sums, gamma, beta, Wu, Wv, Wout,
                                         scale, shift, WuP, WvP, WoutP);
  k_featv<<<B_SEGS / 16, 256, 0, stream>>>(feat, iid, last, bv, scale, shift, WvP, featv);
  k_e<<<(N_NODES / 16 + 7) / 8, 256, 0, stream>>>(feat, iid, seg, we, scale, shift,
                                                  WuP, featv, ebuf, emax);
  k_denom<<<2048, 256, 0, stream>>>(seg, emax, ebuf, denom);
  k_rst<<<512, 256, 0, stream>>>(feat, iid, seg, scale, shift, ebuf, denom, rst);
  k_out<<<(B_SEGS / 16) * 2, 256, 0, stream>>>(rst, WoutP, out);
}
